// TorchReshapedEmbeddingGatherEinsum_78383153152659
// MI455X (gfx1250) — matrix operand order repaired and checked by reading, not compiled, not run on hardware
//
#include <hip/hip_runtime.h>

// Problem constants (from the reference)
#define B_ 4
#define T_ 4096
#define I_ 1024
#define E_ 16
#define K_ 512
#define J_ 1024

// Tiling
#define BM 128
#define BN 128
#define KSTEP 32
#define A_STRIDE 40    // 32 + 8 pad (bf16 elements) -> 80B row stride
#define B_STRIDE 136   // 128 + 8 pad (bf16 elements) -> 272B row stride

typedef __attribute__((ext_vector_type(16))) __bf16 v16bf;
typedef __attribute__((ext_vector_type(8)))  __bf16 v8bf;
typedef __attribute__((ext_vector_type(8)))  float  v8f;

static __device__ __forceinline__ v8bf cvt8(float4 u, float4 v) {
    v8bf r;
    r[0] = (__bf16)u.x; r[1] = (__bf16)u.y; r[2] = (__bf16)u.z; r[3] = (__bf16)u.w;
    r[4] = (__bf16)v.x; r[5] = (__bf16)v.y; r[6] = (__bf16)v.z; r[7] = (__bf16)v.w;
    return r;
}

__global__ __launch_bounds__(256)
void gather_groupgemm_bf16_wmma(const float* __restrict__ X,
                                const long long* __restrict__ ind,
                                const float* __restrict__ W,
                                float* __restrict__ Y)
{
    // Double-buffered tiles: 2*(10240 + 8704) = 37888 B of LDS
    __shared__ __align__(16) __bf16 Abuf[2][BM * A_STRIDE];
    __shared__ __align__(16) __bf16 Bbuf[2][KSTEP * B_STRIDE];

    const int tid  = threadIdx.x;
    const int lane = tid & 31;
    const int wave = tid >> 5;   // 8 waves
    const int wrow = wave & 1;   // 2 wave-rows  (64 rows each)
    const int wcol = wave >> 1;  // 4 wave-cols  (32 cols each)

    const int be = blockIdx.y;           // 0..63
    const int b  = be >> 4;              // / E_
    const int e  = be & 15;              // % E_
    const int m0 = (blockIdx.x & 3) * BM;   // 4 m-tiles
    const int n0 = (blockIdx.x >> 2) * BN;  // 8 n-tiles

    // ---- A loader: each thread owns half (16 fp32) of one gathered row ----
    const int a_m    = tid >> 1;   // 0..127 row within tile
    const int a_half = tid & 1;    // which 16-wide k half
    const long long idx = ind[(long long)(b * E_ + e) * K_ + (m0 + a_m)];
    const float* Xrow = X + ((long long)b * T_ + idx) * I_;

    // ---- B loader: each thread owns 16 fp32 of one W row ----
    const int b_k = tid >> 3;          // 0..31
    const int b_n = (tid & 7) * 16;    // 0,16,...,112
    const float* Wbase = W + ((long long)e * I_ + b_k) * (long long)J_ + (n0 + b_n);

    const int a_dst_off = a_m * A_STRIDE + a_half * 16;
    const int b_dst_off = b_k * B_STRIDE + b_n;

    v8f acc[4][2];
    #pragma unroll
    for (int i = 0; i < 4; ++i)
        #pragma unroll
        for (int j = 0; j < 2; ++j)
            acc[i][j] = (v8f){0.f, 0.f, 0.f, 0.f, 0.f, 0.f, 0.f, 0.f};

    // ---- prologue: stage k0 = 0 into buffer 0 ----
    {
        const float4* asrc = (const float4*)(Xrow + a_half * 16);
        const float4* bsrc = (const float4*)(Wbase);
        float4 a0 = asrc[0], a1 = asrc[1], a2 = asrc[2], a3 = asrc[3];
        float4 w0 = bsrc[0], w1 = bsrc[1], w2 = bsrc[2], w3 = bsrc[3];
        *(v8bf*)(&Abuf[0][a_dst_off])     = cvt8(a0, a1);
        *(v8bf*)(&Abuf[0][a_dst_off + 8]) = cvt8(a2, a3);
        *(v8bf*)(&Bbuf[0][b_dst_off])     = cvt8(w0, w1);
        *(v8bf*)(&Bbuf[0][b_dst_off + 8]) = cvt8(w2, w3);
    }

    const __bf16* Acur  = Abuf[0];
    const __bf16* Bcur  = Bbuf[0];
    __bf16*       Anext = Abuf[1];
    __bf16*       Bnext = Bbuf[1];

    const int l15   = lane & 15;
    const int kbase = (lane < 16) ? 0 : 8;  // A: lanes<16 -> K{0-7,16-23}, else K{8-15,24-31}

    for (int k0 = 0; k0 < I_; k0 += KSTEP) {
        __syncthreads();  // buf[cur] stores visible; buf[next] free to overwrite

        const bool has_next = (k0 + KSTEP) < I_;

        // ---- issue next tile's global loads early (prefetch into regs) ----
        float4 a0, a1, a2, a3, w0, w1, w2, w3;
        if (has_next) {
            const float4* asrc = (const float4*)(Xrow + (k0 + KSTEP) + a_half * 16);
            const float4* bsrc = (const float4*)(Wbase + (long long)(k0 + KSTEP) * J_);
            a0 = asrc[0]; a1 = asrc[1]; a2 = asrc[2]; a3 = asrc[3];
            w0 = bsrc[0]; w1 = bsrc[1]; w2 = bsrc[2]; w3 = bsrc[3];
        }

        // ---- fragments per ISA VGPR layouts (from current buffer) ----
        v16bf afrag[4];
        #pragma unroll
        for (int i = 0; i < 4; ++i) {
            const __bf16* ap = &Acur[(wrow * 64 + i * 16 + l15) * A_STRIDE + kbase];
            v8bf alo = *(const v8bf*)(ap);        // K base..base+7
            v8bf ahi = *(const v8bf*)(ap + 16);   // K base+16..base+23
            afrag[i] = __builtin_shufflevector(alo, ahi,
                        0,1,2,3,4,5,6,7,8,9,10,11,12,13,14,15);
        }
        v16bf bfrag[2];
        #pragma unroll
        for (int j = 0; j < 2; ++j) {
            // B: lane holds N=lane%16; K{kbase..kbase+7, kbase+16..kbase+23}
            const __bf16* bp = &Bcur[kbase * B_STRIDE + (wcol * 32 + j * 16 + l15)];
            #pragma unroll
            for (int s = 0; s < 16; ++s)
                bfrag[j][s] = bp[((s & 7) + (s >> 3) * 16) * B_STRIDE];
        }

        // ---- 8x v_wmma_f32_16x16x32_bf16 per wave per k-step ----
        #pragma unroll
        for (int i = 0; i < 4; ++i)
            #pragma unroll
            for (int j = 0; j < 2; ++j)
                acc[i][j] = __builtin_amdgcn_wmma_f32_16x16x32_bf16(
                    false, afrag[i], false, bfrag[j],
                    (short)0, acc[i][j], false, false);

        // ---- convert + stage next tile into the other buffer ----
        if (has_next) {
            *(v8bf*)(&Anext[a_dst_off])     = cvt8(a0, a1);
            *(v8bf*)(&Anext[a_dst_off + 8]) = cvt8(a2, a3);
            *(v8bf*)(&Bnext[b_dst_off])     = cvt8(w0, w1);
            *(v8bf*)(&Bnext[b_dst_off + 8]) = cvt8(w2, w3);
        }

        // ping-pong
        const __bf16* tA = Acur; Acur = Anext; Anext = (__bf16*)tA;
        const __bf16* tB = Bcur; Bcur = Bnext; Bnext = (__bf16*)tB;
    }

    // ---- epilogue: C/D layout -> Y[b][e][m][n] (fp32) ----
    const int mhi = (lane >> 4) * 8;
    #pragma unroll
    for (int i = 0; i < 4; ++i) {
        #pragma unroll
        for (int j = 0; j < 2; ++j) {
            const int mb = m0 + wrow * 64 + i * 16 + mhi;
            const int nb = n0 + wcol * 32 + j * 16 + l15;
            float* dst = Y + ((long long)(b * E_ + e) * K_ + mb) * (long long)J_ + nb;
            #pragma unroll
            for (int r = 0; r < 8; ++r)
                dst[(long long)r * J_] = acc[i][j][r];
        }
    }
}

extern "C" void kernel_launch(void* const* d_in, const int* in_sizes, int n_in,
                              void* d_out, int out_size, void* d_ws, size_t ws_size,
                              hipStream_t stream) {
    const float*     X   = (const float*)d_in[0];      // [B,T,I] fp32
    const long long* ind = (const long long*)d_in[1];  // [B,E,K] int64
    const float*     W   = (const float*)d_in[2];      // [E,I,J] fp32
    float*           Y   = (float*)d_out;              // [B,E,K,J] fp32

    dim3 grid((K_ / BM) * (J_ / BN), B_ * E_);  // (32, 64)
    dim3 block(256);
    gather_groupgemm_bf16_wmma<<<grid, block, 0, stream>>>(X, ind, W, Y);
}
